// DynamicGATLayer_88364657148210
// MI455X (gfx1250) — matrix operand order repaired and checked by reading, not compiled, not run on hardware
//
#include <hip/hip_runtime.h>
#include <hip/hip_bf16.h>

typedef __attribute__((ext_vector_type(16))) _Float16 v16h;
typedef __attribute__((ext_vector_type(8)))  float    v8f;

#define BB 4
#define NN 2048
#define DD 64
#define HH 4
#define DH 16
#define NEGV (-1000000000.0f)

// A-matrix (16x32 f16) K-base for VGPR-halfpair v, lane-half hi (ISA 7.12.2)
__device__ __forceinline__ int a_kbase(int v, int hi) {
    return ((v < 4) ? 2 * v : 16 + 2 * (v - 4)) + hi * 8;
}

// ---------------------------------------------------------------------------
// Pre-swizzle x (f32) into A-fragment layout, 16x32 tiles:
//   frag[(tile16*2+kk)*512 + lane*16 + vi] = x[row, kk*32+k2]
//   lane = m + hi*16 ; hi = (k2>>3)&1 ; vi = (k2&7) | ((k2>>4)<<3)
__global__ void k_swz_x(const float* __restrict__ s, _Float16* __restrict__ d) {
    int idx = blockIdx.x * 256 + threadIdx.x;        // over B*N*D
    if (idx >= BB * NN * DD) return;
    int g = idx >> 6, c = idx & 63;
    int it = g >> 4, m = g & 15;
    int kk = c >> 5, k2 = c & 31;
    int hi = (k2 >> 3) & 1;
    int vi = (k2 & 7) | ((k2 >> 4) << 3);
    int lane = m + (hi << 4);
    d[(((size_t)it * 2 + kk) << 9) + (lane << 4) + vi] = (_Float16)s[idx];
}

// Pre-swizzle a 64x64 weight W (used as B with B(k,n)=W[n,k]) into
// B-fragment layout: frag[(kk*4+t)*512 + lane*16 + ((k2&7)|((k2>>4)<<3))],
//   lane = n + ((k2>>3)&1)*16, t = n_/16, kk = k/32.
__global__ void k_swz_w(const float* __restrict__ s, _Float16* __restrict__ d) {
    int idx = blockIdx.x * 256 + threadIdx.x;        // over 64*64
    if (idx >= DD * DD) return;
    int n_ = idx >> 6, k = idx & 63;                 // element W[n_, k]
    int t = n_ >> 4, n = n_ & 15;
    int kk = k >> 5, k2 = k & 31;
    int hi = (k2 >> 3) & 1, vi = (k2 & 7) | ((k2 >> 4) << 3);
    int lane = n + (hi << 4);
    d[(((size_t)kk * 4 + t) << 9) + (lane << 4) + vi] = (_Float16)s[idx];
}

// ------------------------------------------------- projections + e_i / e_j
// grid.x = (B*N)/16 row-tiles, grid.y = 0 (src) / 1 (tgt). 1 wave per block.
// All WMMA operands arrive as single v16h global loads from swizzled buffers.
__global__ __launch_bounds__(32)
void k_proj(const _Float16* __restrict__ xa,
            const _Float16* __restrict__ wsrc_s,
            const _Float16* __restrict__ wtgt_s,
            const float* __restrict__ attn,
            float* __restrict__ ei, float* __restrict__ ej,
            _Float16* __restrict__ htswz) {
    const int which = blockIdx.y;
    const _Float16* W = which ? wtgt_s : wsrc_s;
    const int lane = threadIdx.x;
    const int m = lane & 15, hi = lane >> 4;
    const int it = blockIdx.x;

    __shared__ __align__(16) float sc[16 * DD];      // 4 KB

    v8f acc[4] = {};
    for (int kk = 0; kk < 2; ++kk) {
        v16h A = *(const v16h*)(xa + (((size_t)it * 2 + kk) << 9) + (lane << 4));
        for (int t = 0; t < 4; ++t) {
            v16h Bf = *(const v16h*)(W + (((size_t)kk * 4 + t) << 9) + (lane << 4));
            acc[t] = __builtin_amdgcn_wmma_f32_16x16x32_f16(
                false, A, false, Bf, (short)0, acc[t], false, false);
        }
    }
    for (int t = 0; t < 4; ++t)
        for (int r = 0; r < 8; ++r)
            sc[(r + hi * 8) * DD + t * 16 + m] = acc[t][r];
    __syncthreads();

    if (lane < 16) {
        int g = it * 16 + lane;                      // g = b*N + row
        float* eo = which ? ej : ei;
        int acol = which ? DH : 0;
        for (int h = 0; h < HH; ++h) {
            float dot = 0.f;
            for (int d = 0; d < DH; ++d)
                dot += sc[lane * DD + h * DH + d] * attn[h * (2 * DH + 1) + acol + d];
            eo[g * HH + h] = dot;
        }
        if (which) {
            // write h_tgt directly in B-fragment layout for the aggregation:
            // per (b, j-tile, h): 512 halves, lane2*16 + ((k&7)|((k>>4)<<3))
            int b = g >> 11, j = g & (NN - 1);
            int jt = j >> 5, k = j & 31;
            size_t base = ((size_t)(b * (NN / 32) + jt) * HH) << 9;
            for (int d = 0; d < DD; ++d) {
                int h = d >> 4, n = d & 15;
                int l2 = n + (((k >> 3) & 1) << 4);
                htswz[base + ((size_t)h << 9) + (l2 << 4) + ((k & 7) | ((k >> 4) << 3))] =
                    (_Float16)sc[lane * DD + d];
            }
        }
    }
}

// ---------------------------------------------- streaming softmax row stats
__global__ __launch_bounds__(32)
void k_stats(const int* __restrict__ adj, const float* __restrict__ cong,
             const float* __restrict__ ei, const float* __restrict__ ej,
             const float* __restrict__ attn,
             float* __restrict__ mrow, float* __restrict__ ilrow) {
    const int g = blockIdx.x;                        // b*N + i
    const int b = g >> 11, i = g & (NN - 1);
    const int lane = threadIdx.x;
    const float ci = cong[g];
    float eih[HH], ach[HH], m_[HH], l_[HH];
    for (int h = 0; h < HH; ++h) {
        eih[h] = ei[g * HH + h];
        ach[h] = attn[h * (2 * DH + 1) + 2 * DH];
        m_[h] = -1.0e30f; l_[h] = 0.f;
    }
    const int* arow = adj + (size_t)i * NN;
    const float* cb = cong + (size_t)b * NN;
    const float* ejb = ej + (size_t)b * NN * HH;
    for (int j = lane; j < NN; j += 32) {
        int a = arow[j];
        float cd = fabsf(ci - cb[j]);
        for (int h = 0; h < HH; ++h) {
            float e = eih[h] + ejb[j * HH + h] + cd * ach[h];
            e = (e > 0.f) ? e : 0.2f * e;
            e = a ? e : NEGV;
            float mn = fmaxf(m_[h], e);
            l_[h] = l_[h] * __expf(m_[h] - mn) + __expf(e - mn);
            m_[h] = mn;
        }
    }
    for (int off = 16; off > 0; off >>= 1) {
        for (int h = 0; h < HH; ++h) {
            float mo = __shfl_xor(m_[h], off);
            float lo = __shfl_xor(l_[h], off);
            float mn = fmaxf(m_[h], mo);
            l_[h] = l_[h] * __expf(m_[h] - mn) + lo * __expf(mo - mn);
            m_[h] = mn;
        }
    }
    if (lane == 0)
        for (int h = 0; h < HH; ++h) {
            mrow[g * HH + h] = m_[h];
            ilrow[g * HH + h] = 1.0f / l_[h];
        }
}

// ------------- aggregation (WMMA) + W_out GEMM (WMMA) + residual + layernorm
// grid.x = B*(N/16); 1 wave per 16-row i-tile. adj tile staged with
// GLOBAL_LOAD_ASYNC_TO_LDS_B128 (ASYNCcnt); h_tgt/W_out fragments are direct
// v16h global loads from the pre-swizzled buffers.
__global__ __launch_bounds__(32)
void k_aggr(const int* __restrict__ adj, const float* __restrict__ cong,
            const float* __restrict__ ei, const float* __restrict__ ej,
            const float* __restrict__ attn,
            const float* __restrict__ mrow, const float* __restrict__ ilrow,
            const _Float16* __restrict__ htswz, const _Float16* __restrict__ wout_s,
            const float* __restrict__ x, const float* __restrict__ bout,
            const float* __restrict__ gmm, const float* __restrict__ bta,
            float* __restrict__ out) {
    const int gt = blockIdx.x;
    const int b = gt / (NN / 16);
    const int i0 = (gt % (NN / 16)) * 16;
    const int lane = threadIdx.x;
    const int m = lane & 15, hi = lane >> 4;

    __shared__ __align__(16) int   adjt[16 * 32];    // 2 KB
    __shared__ __align__(16) float cjs[32];
    __shared__ __align__(16) float ejs[32 * HH];
    __shared__ __align__(16) float sc[16 * DD];      // 4 KB
    __shared__ float ci_s[16], eis[16 * HH], ms[16 * HH], ils[16 * HH];

    if (lane < 16) {
        int gi = b * NN + i0 + lane;
        ci_s[lane] = cong[gi];
        for (int h = 0; h < HH; ++h) {
            eis[lane * HH + h] = ei[gi * HH + h];
            ms[lane * HH + h]  = mrow[gi * HH + h];
            ils[lane * HH + h] = ilrow[gi * HH + h];
        }
    }
    __syncthreads();
    float cir = ci_s[m];
    float eir[HH], mr[HH], ilr[HH], ach[HH];
    for (int h = 0; h < HH; ++h) {
        eir[h] = eis[m * HH + h];
        mr[h]  = ms[m * HH + h];
        ilr[h] = ils[m * HH + h];
        ach[h] = attn[h * (2 * DH + 1) + 2 * DH];
    }

    v8f acc[HH] = {};
    for (int j0 = 0; j0 < NN; j0 += 32) {
        __syncthreads();
        // adj tile 16x32 ints via async b128 copies to LDS (4 per lane)
        for (int it2 = 0; it2 < 4; ++it2) {
            int idx = lane + it2 * 32;               // 128 chunks of 16B
            int r = idx >> 3;
            int col = (idx & 7) * 4;
            unsigned ldsa = (unsigned)(uintptr_t)(&adjt[idx * 4]);
            unsigned long long ga =
                (unsigned long long)(uintptr_t)(adj + (size_t)(i0 + r) * NN + j0 + col);
            asm volatile("global_load_async_to_lds_b128 %0, %1, off"
                         :: "v"(ldsa), "v"(ga) : "memory");
        }
        cjs[lane] = cong[b * NN + j0 + lane];
        for (int idx = lane; idx < 32 * HH; idx += 32)
            ejs[idx] = ej[(size_t)(b * NN + j0 + (idx >> 2)) * HH + (idx & 3)];
        if (j0 + 32 < NN)
            __builtin_prefetch(
                htswz + (((size_t)(b * (NN / 32) + ((j0 >> 5) + 1)) * HH) << 9), 0, 0);
        asm volatile("s_wait_asynccnt 0x0" ::: "memory");
        __syncthreads();

        size_t hbase = ((size_t)(b * (NN / 32) + (j0 >> 5)) * HH) << 9;
        for (int h = 0; h < HH; ++h) {
            v16h A;
            for (int v = 0; v < 8; ++v) {
                int kb = a_kbase(v, hi);
                for (int s = 0; s < 2; ++s) {
                    int k = kb + s;
                    float e = eir[h] + ejs[k * HH + h] + fabsf(cir - cjs[k]) * ach[h];
                    e = (e > 0.f) ? e : 0.2f * e;
                    e = adjt[m * 32 + k] ? e : NEGV;
                    float w = __expf(e - mr[h]) * ilr[h];
                    A[2 * v + s] = (_Float16)w;
                }
            }
            v16h Bf = *(const v16h*)(htswz + hbase + ((size_t)h << 9) + (lane << 4));
            acc[h] = __builtin_amdgcn_wmma_f32_16x16x32_f16(
                false, A, false, Bf, (short)0, acc[h], false, false);
        }
    }
    __syncthreads();
    for (int h = 0; h < HH; ++h)
        for (int r = 0; r < 8; ++r)
            sc[(r + hi * 8) * DD + h * DH + m] = acc[h][r];
    __syncthreads();

    // out @ W_out.T with pre-swizzled B fragments (single v16h loads)
    v8f acc2[4] = {};
    for (int kk = 0; kk < 2; ++kk) {
        v16h A;
        for (int v = 0; v < 8; ++v) {
            int kb = a_kbase(v, hi) + kk * 32;
            A[2 * v]     = (_Float16)sc[m * DD + kb];
            A[2 * v + 1] = (_Float16)sc[m * DD + kb + 1];
        }
        for (int t = 0; t < 4; ++t) {
            v16h Bf = *(const v16h*)(wout_s + (((size_t)kk * 4 + t) << 9) + (lane << 4));
            acc2[t] = __builtin_amdgcn_wmma_f32_16x16x32_f16(
                false, A, false, Bf, (short)0, acc2[t], false, false);
        }
    }
    __syncthreads();                                 // all reads of sc done
    for (int t = 0; t < 4; ++t)
        for (int r = 0; r < 8; ++r)
            sc[(r + hi * 8) * DD + t * 16 + m] = acc2[t][r];
    __syncthreads();

    if (lane < 16) {
        int gi = b * NN + i0 + lane;
        float s1 = 0.f;
        for (int d = 0; d < DD; ++d) {
            float y = sc[lane * DD + d] + bout[d] + x[(size_t)gi * DD + d];
            sc[lane * DD + d] = y;
            s1 += y;
        }
        float mu = s1 * (1.0f / DD);
        float s2 = 0.f;
        for (int d = 0; d < DD; ++d) {
            float t = sc[lane * DD + d] - mu;
            s2 += t * t;
        }
        float inv = rsqrtf(s2 * (1.0f / DD) + 1e-5f);
        for (int d = 0; d < DD; ++d)
            out[(size_t)gi * DD + d] =
                (sc[lane * DD + d] - mu) * inv * gmm[d] + bta[d];
    }
}

// --------------------------------------------------------------------- host
extern "C" void kernel_launch(void* const* d_in, const int* in_sizes, int n_in,
                              void* d_out, int out_size, void* d_ws, size_t ws_size,
                              hipStream_t stream) {
    const float* x     = (const float*)d_in[0];
    const int*   adj   = (const int*)d_in[1];
    const float* cong  = (const float*)d_in[2];
    const float* Wsrc  = (const float*)d_in[3];
    const float* Wtgt  = (const float*)d_in[4];
    const float* attn  = (const float*)d_in[5];
    const float* Wout  = (const float*)d_in[6];
    const float* bout  = (const float*)d_in[7];
    const float* gmm   = (const float*)d_in[8];
    const float* bta   = (const float*)d_in[9];
    float* out = (float*)d_out;

    char* ws = (char*)d_ws;
    size_t off = 0;
    auto carve = [&](size_t bytes) {
        char* p = ws + off;
        off += bytes;
        off = (off + 255) & ~(size_t)255;
        return p;
    };
    const int NX = BB * NN * DD;                     // 524288
    _Float16* xa     = (_Float16*)carve((size_t)NX * 2);
    _Float16* wsrc_s = (_Float16*)carve(DD * DD * 2);
    _Float16* wtgt_s = (_Float16*)carve(DD * DD * 2);
    _Float16* wout_s = (_Float16*)carve(DD * DD * 2);
    float*    ei     = (float*)carve((size_t)BB * NN * HH * 4);
    float*    ej     = (float*)carve((size_t)BB * NN * HH * 4);
    float*    mrow   = (float*)carve((size_t)BB * NN * HH * 4);
    float*    ilrow  = (float*)carve((size_t)BB * NN * HH * 4);
    _Float16* htswz  = (_Float16*)carve((size_t)NX * 2);

    k_swz_x<<<(NX + 255) / 256, 256, 0, stream>>>(x, xa);
    k_swz_w<<<(DD * DD + 255) / 256, 256, 0, stream>>>(Wsrc, wsrc_s);
    k_swz_w<<<(DD * DD + 255) / 256, 256, 0, stream>>>(Wtgt, wtgt_s);
    k_swz_w<<<(DD * DD + 255) / 256, 256, 0, stream>>>(Wout, wout_s);

    k_proj<<<dim3(BB * NN / 16, 2), 32, 0, stream>>>(xa, wsrc_s, wtgt_s, attn,
                                                     ei, ej, htswz);
    k_stats<<<BB * NN, 32, 0, stream>>>(adj, cong, ei, ej, attn, mrow, ilrow);
    k_aggr<<<BB * (NN / 16), 32, 0, stream>>>(adj, cong, ei, ej, attn, mrow, ilrow,
                                              htswz, wout_s, x, bout, gmm, bta, out);
    (void)in_sizes; (void)n_in; (void)out_size; (void)ws_size;
}
